// Mamba_6940667150618
// MI455X (gfx1250) — matrix operand order repaired and checked by reading, not compiled, not run on hardware
//
#include <hip/hip_runtime.h>

// ---------------------------------------------------------------------------
// Mamba forward for MI455X (gfx1250, wave32, WMMA + async Global->LDS).
// Pipeline: cvt(f32->bf16) -> WMMA in_proj -> conv+SiLU -> WMMA x_proj ->
//           WMMA dt_proj(+softplus) -> fused selective scan (+gate) ->
//           WMMA out_proj.
// GEMMs: v_wmma_f32_16x16x32_bf16, tiles staged with
// global_load_async_to_lds_b128 (ASYNCcnt) into double-buffered LDS,
// pipeline unrolled x2 so buffer indices are compile-time constants.
// ---------------------------------------------------------------------------

typedef __bf16 bf16;
typedef __attribute__((ext_vector_type(16))) __bf16 v16bf;
typedef __attribute__((ext_vector_type(8)))  __bf16 v8bf;
typedef __attribute__((ext_vector_type(8)))  float   v8f;
typedef __attribute__((ext_vector_type(4)))  unsigned int u32x4;

constexpr int Bb    = 2;
constexpr int Ls    = 1024;
constexpr int Dm    = 1024;
constexpr int EDc   = 2048;          // d_inner
constexpr int Ns    = 16;            // d_state
constexpr int Kc    = 4;             // d_conv
constexpr int Rr    = 64;            // dt_rank
constexpr int Mrows = Bb * Ls;       // 2048
constexpr int Fdim  = Rr + 2 * Ns;   // 96

__device__ __forceinline__ bf16 f2bf(float f) {
  unsigned u = __builtin_bit_cast(unsigned, f);
  u += 0x7fffu + ((u >> 16) & 1u);               // round-to-nearest-even
  unsigned short h = (unsigned short)(u >> 16);
  return __builtin_bit_cast(bf16, h);
}

__device__ __forceinline__ float softplusf(float x) {
  return (x > 20.f) ? x : log1pf(__expf(x));
}

// low 32 bits of a generic pointer to LDS == addrspace(3) byte offset
__device__ __forceinline__ unsigned lds_off_of(const void* p) {
  return (unsigned)(unsigned long long)p;
}

// async DMA: 16 bytes global -> LDS, tracked by ASYNCcnt (CDNA5)
__device__ __forceinline__ void async_b128(unsigned lds, const void* g) {
  asm volatile("global_load_async_to_lds_b128 %0, %1, off"
               :: "v"(lds), "v"(g) : "memory");
}

__device__ __forceinline__ void wait_async0() {
  asm volatile("s_wait_asynccnt 0x0" ::: "memory");
}

// ---------------------------------------------------------------------------
// f32 -> bf16 conversion (grid-stride)
// ---------------------------------------------------------------------------
__global__ void cvt_bf16_kernel(const float* __restrict__ src,
                                bf16* __restrict__ dst, int n) {
  int i = blockIdx.x * blockDim.x + threadIdx.x;
  int stride = gridDim.x * blockDim.x;
  for (; i < n; i += stride) dst[i] = f2bf(src[i]);
}

// ---------------------------------------------------------------------------
// WMMA GEMM:  C[M,N] = A[M,K] * Bw[N,K]^T   (TN layout, bf16 in, f32 out)
// Block tile 128x64, 8 waves (4x2), each wave 32x32 = 2x2 WMMA 16x16 tiles.
// Tiles staged via async global->LDS, double buffered, 1 barrier / k-step,
// pipeline unrolled by 2 (requires K % 64 == 0; M % 128 == 0; N ragged OK).
// act: 0 = none, 1 = softplus(x + bias[col])
// ---------------------------------------------------------------------------
union FragAB { v16bf v; v8bf h[2]; };

__global__ __launch_bounds__(256)
void wmma_gemm_kernel(const bf16* __restrict__ A, const bf16* __restrict__ Bw,
                      float* __restrict__ C, const float* __restrict__ bias,
                      int M, int N, int Kd, int act) {
  __shared__ __align__(16) bf16 As[2][128 * 32];  // 2 x 8 KB
  __shared__ __align__(16) bf16 Bs[2][64 * 32];   // 2 x 4 KB
  const int tid = threadIdx.x;
  const int m0 = blockIdx.y * 128;
  const int n0 = blockIdx.x * 64;

  // staging assignment: each thread owns 32 B of A tile and 16 B of B tile
  const int arow = tid >> 1, ahalf = tid & 1;   // 128 rows x 2 halves of 16
  const int brow = tid >> 2, bq    = tid & 3;   // 64 rows x 4 chunks of 8
  const bf16* Ap = A + (size_t)(m0 + arow) * Kd + ahalf * 16;
  const bf16* Bp = Bw + (size_t)(n0 + brow) * Kd + bq * 8;
  const bool bok = (n0 + brow) < N;

  const unsigned aoff[2] = { lds_off_of(&As[0][arow * 32 + ahalf * 16]),
                             lds_off_of(&As[1][arow * 32 + ahalf * 16]) };
  const unsigned boff[2] = { lds_off_of(&Bs[0][brow * 32 + bq * 8]),
                             lds_off_of(&Bs[1][brow * 32 + bq * 8]) };

  // zero-fill OOB B rows once; those LDS regions are never async-written
  if (!bok) {
    const u32x4 z4 = {0u, 0u, 0u, 0u};
    *(u32x4*)&Bs[0][brow * 32 + bq * 8] = z4;
    *(u32x4*)&Bs[1][brow * 32 + bq * 8] = z4;
  }

  auto issue = [&](int ks, int buf) {
    const bf16* ga = Ap + (size_t)ks * 32;
    async_b128(aoff[buf], ga);                 // K 0..7   of this half
    async_b128(aoff[buf] + 16, ga + 8);        // K 8..15  of this half
    if (bok) async_b128(boff[buf], Bp + (size_t)ks * 32);
  };

  // wave / lane tiling
  const int w = tid >> 5, lane = tid & 31;
  const int wm = (w >> 1) * 32, wn = (w & 1) * 32;
  const int lr = lane & 15, lh = lane >> 4;

  v8f acc[2][2] = {};
  const int nk = Kd >> 5;   // even (K % 64 == 0)

  // one pipeline stage: drain own async loads, barrier, prefetch tile nks
  // into buffer buf^1, run 2x2 WMMAs from buffer buf (buf is a literal).
  auto stage = [&](int buf, int nks) {
    wait_async0();            // this wave's async tile loads have landed
    __syncthreads();          // everyone's landed; other buffer is free
    if (nks < nk) issue(nks, buf ^ 1);

    // Build A 16x32 / B 32x16 fragments in contiguous k order (k = 16*lh + s), same for A and B
    FragAB fa[2]; v16bf fb[2];
#pragma unroll
    for (int fi = 0; fi < 2; ++fi) {
      const bf16* p = &As[buf][(wm + fi * 16 + lr) * 32 + lh * 16];
      fa[fi].h[0] = *(const v8bf*)(p);        // K 0..7   (or 16..23)
      fa[fi].h[1] = *(const v8bf*)(p + 8);    // K 8..15  (or 24..31)
    }
#pragma unroll
    for (int fj = 0; fj < 2; ++fj)
      fb[fj] = *(const v16bf*)&Bs[buf][(wn + fj * 16 + lr) * 32 + lh * 16];
#pragma unroll
    for (int fi = 0; fi < 2; ++fi)
#pragma unroll
      for (int fj = 0; fj < 2; ++fj)
        acc[fi][fj] = __builtin_amdgcn_wmma_f32_16x16x32_bf16(
            false, fa[fi].v, false, fb[fj], (short)0, acc[fi][fj],
            false, false);
  };

  issue(0, 0);
  for (int ks = 0; ks < nk; ks += 2) {
    stage(0, ks + 1);   // compute buf0, prefetch -> buf1
    stage(1, ks + 2);   // compute buf1, prefetch -> buf0
  }

  // epilogue: C/D layout — VGPR r: row = base + lh*8 + r, col = base + lr
#pragma unroll
  for (int fi = 0; fi < 2; ++fi)
#pragma unroll
    for (int fj = 0; fj < 2; ++fj) {
      const int col = n0 + wn + fj * 16 + lr;
      if (col < N) {
        const float bcol = (act == 1) ? bias[col] : 0.f;
#pragma unroll
        for (int r = 0; r < 8; ++r) {
          const int row = m0 + wm + fi * 16 + lh * 8 + r;
          float v = acc[fi][fj][r];
          if (act == 1) v = softplusf(v + bcol);
          C[(size_t)row * N + col] = v;
        }
      }
    }
}

// ---------------------------------------------------------------------------
// depthwise causal conv1d + bias + SiLU ; writes u in f32 (scan) + bf16 (GEMM)
// xz: [Mrows, 2*EDc] (xh = cols 0..EDc-1)
// ---------------------------------------------------------------------------
__global__ void conv_silu_kernel(const float* __restrict__ xz,
                                 const float* __restrict__ cw,
                                 const float* __restrict__ cb,
                                 float* __restrict__ uf,
                                 bf16* __restrict__ ubf) {
  const int idx = blockIdx.x * blockDim.x + threadIdx.x;  // Mrows*EDc
  const int e = idx & (EDc - 1);
  const int row = idx >> 11;           // b*Ls + t
  const int t = row & (Ls - 1);
  float acc = cb[e];
#pragma unroll
  for (int k = 0; k < Kc; ++k) {
    const int tt = t + k - (Kc - 1);
    if (tt >= 0)
      acc = fmaf(xz[(size_t)(row + k - (Kc - 1)) * (2 * EDc) + e],
                 cw[e * Kc + k], acc);
  }
  const float s = acc / (1.f + __expf(-acc));   // SiLU
  uf[idx] = s;
  ubf[idx] = f2bf(s);
}

// ---------------------------------------------------------------------------
// extract dt_low columns (0..63) of dBC as bf16 for the dt_proj GEMM
// ---------------------------------------------------------------------------
__global__ void extract_dtlow_kernel(const float* __restrict__ dBC,
                                     bf16* __restrict__ o) {
  const int i = blockIdx.x * blockDim.x + threadIdx.x;  // Mrows*Rr
  const int m = i >> 6, r = i & 63;
  o[i] = f2bf(dBC[(size_t)m * Fdim + r]);
}

// ---------------------------------------------------------------------------
// fused selective scan + skip (u*D) + SiLU(z) gate -> yg (bf16 for out_proj)
// One thread owns one (b, e) channel: 16 states in registers, sequential in t.
// B_t / C_t broadcast via LDS in chunks of 8 timesteps.
// grid = Bb * (EDc/64), block = 64
// ---------------------------------------------------------------------------
__global__ void scan_kernel(const float* __restrict__ dBC,
                            const float* __restrict__ delta,
                            const float* __restrict__ uf,
                            const float* __restrict__ xz,
                            const float* __restrict__ A_log,
                            const float* __restrict__ Dp,
                            bf16* __restrict__ yg) {
  constexpr int TCH = 8;
  const int blocksPerBatch = EDc / 64;
  const int b = blockIdx.x / blocksPerBatch;
  const int e = (blockIdx.x % blocksPerBatch) * 64 + threadIdx.x;
  __shared__ float sB[TCH][Ns];
  __shared__ float sC[TCH][Ns];

  float Arow[Ns];
#pragma unroll
  for (int n = 0; n < Ns; ++n) Arow[n] = -__expf(A_log[e * Ns + n]);
  const float dpe = Dp[e];
  float h[Ns];
#pragma unroll
  for (int n = 0; n < Ns; ++n) h[n] = 0.f;

  for (int t0 = 0; t0 < Ls; t0 += TCH) {
    __syncthreads();
    for (int i = threadIdx.x; i < TCH * 2 * Ns; i += 64) {
      const int tt = i >> 5, c = i & 31;
      const float v = dBC[(size_t)(b * Ls + t0 + tt) * Fdim + Rr + c];
      if (c < Ns) sB[tt][c] = v; else sC[tt][c - Ns] = v;
    }
    __syncthreads();
#pragma unroll
    for (int ti = 0; ti < TCH; ++ti) {
      const size_t row = (size_t)(b * Ls + t0 + ti);
      const float dlt = delta[row * EDc + e];
      const float uu  = uf[row * EDc + e];
      const float zz  = xz[row * (2 * EDc) + EDc + e];
      const float du = dlt * uu;
      float y = 0.f;
#pragma unroll
      for (int n = 0; n < Ns; ++n) {
        const float dA = __expf(dlt * Arow[n]);      // exp(delta*A)
        h[n] = fmaf(dA, h[n], du * sB[ti][n]);       // h = dA*h + dBu
        y = fmaf(h[n], sC[ti][n], y);                // y += h*C
      }
      y = fmaf(uu, dpe, y);                          // + u*D
      const float g = zz / (1.f + __expf(-zz));      // SiLU(z) gate
      yg[row * EDc + e] = f2bf(y * g);
    }
  }
}

// ---------------------------------------------------------------------------
// host launcher
// ---------------------------------------------------------------------------
extern "C" void kernel_launch(void* const* d_in, const int* in_sizes, int n_in,
                              void* d_out, int out_size, void* d_ws,
                              size_t ws_size, hipStream_t stream) {
  const float* x      = (const float*)d_in[0];  // [B,L,D]
  const float* W_in   = (const float*)d_in[1];  // [2*ED, D]
  const float* conv_w = (const float*)d_in[2];  // [ED,1,K]
  const float* conv_b = (const float*)d_in[3];  // [ED]
  const float* W_xp   = (const float*)d_in[4];  // [96, ED]
  const float* W_dt   = (const float*)d_in[5];  // [ED, 64]
  const float* dt_b   = (const float*)d_in[6];  // [ED]
  const float* A_log  = (const float*)d_in[7];  // [ED, N]
  const float* D_par  = (const float*)d_in[8];  // [ED]
  const float* W_out  = (const float*)d_in[9];  // [D, ED]
  float* out = (float*)d_out;                   // [B,L,D]

  size_t off = 0;
  auto alloc = [&](size_t bytes) -> void* {
    void* p = (char*)d_ws + off;
    off += (bytes + 255) & ~(size_t)255;
    return p;
  };
  bf16*  x_bf    = (bf16*) alloc((size_t)Mrows * Dm * 2);
  bf16*  Win_bf  = (bf16*) alloc((size_t)2 * EDc * Dm * 2);
  float* xz      = (float*)alloc((size_t)Mrows * 2 * EDc * 4);
  float* uf      = (float*)alloc((size_t)Mrows * EDc * 4);
  bf16*  u_bf    = (bf16*) alloc((size_t)Mrows * EDc * 2);
  bf16*  Wxp_bf  = (bf16*) alloc((size_t)Fdim * EDc * 2);
  float* dBC     = (float*)alloc((size_t)Mrows * Fdim * 4);
  bf16*  dtl_bf  = (bf16*) alloc((size_t)Mrows * Rr * 2);
  bf16*  Wdt_bf  = (bf16*) alloc((size_t)EDc * Rr * 2);
  float* delta   = (float*)alloc((size_t)Mrows * EDc * 4);
  bf16*  yg_bf   = (bf16*) alloc((size_t)Mrows * EDc * 2);
  bf16*  Wout_bf = (bf16*) alloc((size_t)Dm * EDc * 2);

  auto cvt = [&](const float* s, bf16* d, int n) {
    cvt_bf16_kernel<<<(n + 255) / 256, 256, 0, stream>>>(s, d, n);
  };
  cvt(x,     x_bf,    Mrows * Dm);
  cvt(W_in,  Win_bf,  2 * EDc * Dm);
  cvt(W_xp,  Wxp_bf,  Fdim * EDc);
  cvt(W_dt,  Wdt_bf,  EDc * Rr);
  cvt(W_out, Wout_bf, Dm * EDc);

  // 1) in_proj: xz = x @ W_in^T   [2048,4096]
  wmma_gemm_kernel<<<dim3((2 * EDc) / 64, Mrows / 128), 256, 0, stream>>>(
      x_bf, Win_bf, xz, nullptr, Mrows, 2 * EDc, Dm, 0);

  // 2) depthwise conv + SiLU -> u
  conv_silu_kernel<<<(Mrows * EDc) / 256, 256, 0, stream>>>(
      xz, conv_w, conv_b, uf, u_bf);

  // 3) x_proj: dBC = u @ W_xp^T   [2048,96]
  wmma_gemm_kernel<<<dim3((Fdim + 63) / 64, Mrows / 128), 256, 0, stream>>>(
      u_bf, Wxp_bf, dBC, nullptr, Mrows, Fdim, EDc, 0);

  // 4) dt_low slice -> bf16
  extract_dtlow_kernel<<<(Mrows * Rr) / 256, 256, 0, stream>>>(dBC, dtl_bf);

  // 5) dt_proj + softplus(+bias): delta   [2048,2048]
  wmma_gemm_kernel<<<dim3(EDc / 64, Mrows / 128), 256, 0, stream>>>(
      dtl_bf, Wdt_bf, delta, dt_b, Mrows, EDc, Rr, 1);

  // 6) fused selective scan + skip + gate -> yg
  scan_kernel<<<Bb * (EDc / 64), 64, 0, stream>>>(
      dBC, delta, uf, xz, A_log, D_par, yg_bf);

  // 7) out_proj: out = yg @ W_out^T   [2048,1024]
  wmma_gemm_kernel<<<dim3(Dm / 64, Mrows / 128), 256, 0, stream>>>(
      yg_bf, Wout_bf, out, nullptr, Mrows, Dm, EDc, 0);
}
